// MSTParserLSTMModel_3547642986900
// MI455X (gfx1250) — compile-verified, matrix operand order checked by reading, not run on hardware
//
#include <hip/hip_runtime.h>
#include <hip/hip_bf16.h>

// ---------------------------------------------------------------------------
// MSTParser BiLSTM + pairwise scorer for MI455X (gfx1250).
// bf16 WMMA GEMMs with pre-packed B fragments (contiguous 32B/lane loads),
// f32 recurrence, fused tanh->LDS(bf16)->WMMA relation scorer.
// ---------------------------------------------------------------------------

#define N_SEQ 512
#define WD    256
#define LD    256
#define FEAT  512   // 2*LD
#define HID   512
#define NREL  100
#define NRELP 112   // padded to multiple of 16 for WMMA N tiles
#define GATES 1024  // 4*LD

typedef __attribute__((ext_vector_type(16))) __bf16 v16bf;
typedef __attribute__((ext_vector_type(8)))  __bf16 v8bf;
typedef __attribute__((ext_vector_type(4)))  __bf16 v4bf;
typedef __attribute__((ext_vector_type(8)))  float  v8f;

// round-to-nearest-even f32 -> bf16
__device__ __forceinline__ __bf16 f2bf(float f) {
  unsigned u = __float_as_uint(f);
  unsigned r = u + 0x7FFFu + ((u >> 16) & 1u);
  unsigned short h = (unsigned short)(r >> 16);
  __bf16 o;
  __builtin_memcpy(&o, &h, 2);
  return o;
}

__device__ __forceinline__ float sigm(float x) {
  return 1.0f / (1.0f + __expf(-x));
}

// gfx1250 has a TANH transcendental; use it if the builtin exists, else a
// cheap v_exp_f32-based formulation (no libm range-handling overhead).
__device__ __forceinline__ float fast_tanh(float x) {
#if __has_builtin(__builtin_amdgcn_tanhf)
  return __builtin_amdgcn_tanhf(x);
#else
  float e = __expf(2.0f * x);
  return 1.0f - 2.0f / (e + 1.0f);
#endif
}

// ---- A fragment (row-major M x K, ISA 7.12.2 16-bit 16x32 layout) ----------
// Per lane the 16 values are two contiguous 8-element runs:
//   k in [kb0, kb0+8) and [kb0+16, kb0+24),  kb0 = k0 + (lane>=16 ? 8 : 0)
// -> two 128-bit loads per lane.
__device__ __forceinline__ v16bf load_a_rm(const __bf16* A, int lda, int row0, int k0) {
  const int lane = (int)(threadIdx.x & 31);
  const int r    = row0 + (lane & 15);
  const int kb0  = k0 + ((lane >> 4) << 3);
  const __bf16* p = A + (size_t)r * lda + kb0;
  v8bf lo = *reinterpret_cast<const v8bf*>(p);        // k: kb0 .. kb0+7
  v8bf hi = *reinterpret_cast<const v8bf*>(p + 16);   // k: kb0+16 .. kb0+23
  v16bf f;
#pragma unroll
  for (int e = 0; e < 8; ++e) { f[e] = lo[e]; f[8 + e] = hi[e]; }
  return f;
}

// ---- B fragment from pre-packed layout -------------------------------------
// Bp layout: [kt][ntGlobal][lane][16 bf16 contiguous]; one 32B read per lane.
__device__ __forceinline__ v16bf load_b_packed(const __bf16* Bp, int numNT,
                                               int kt, int ntGlobal) {
  const int lane = (int)(threadIdx.x & 31);
  const v16bf* p = reinterpret_cast<const v16bf*>(Bp) +
                   ((size_t)(kt * numNT + ntGlobal) * 32 + lane);
  return *p;
}

// ---------------------------------------------------------------------------
// Pack a row-major f32 B (K x Nsrc) into bf16 WMMA fragment order (K x N,
// N padded with zeros).  dst element i = [kt][nt][lane][e].
// ---------------------------------------------------------------------------
__global__ void pack_b_kernel(const float* __restrict__ src, __bf16* __restrict__ dst,
                              int K, int N, int Nsrc) {
  int i = blockIdx.x * blockDim.x + threadIdx.x;
  if (i >= K * N) return;
  int e     = i & 15;
  int lane  = (i >> 4) & 31;
  int t     = i >> 9;              // kt * (N/16) + nt
  int numNT = N >> 4;
  int kt = t / numNT, nt = t - kt * numNT;
  int v = e >> 1, p = e & 1;
  int klocal = ((lane >> 4) << 3) + ((v & 4) << 2) + ((v & 3) << 1) + p;
  int k = kt * 32 + klocal;
  int n = nt * 16 + (lane & 15);
  float val = (n < Nsrc) ? src[(size_t)k * Nsrc + n] : 0.0f;
  dst[i] = f2bf(val);
}

__global__ void embed_kernel(const int* __restrict__ tok, const float* __restrict__ E,
                             __bf16* __restrict__ xb) {
  int idx = blockIdx.x * blockDim.x + threadIdx.x;   // over N_SEQ*WD
  if (idx < N_SEQ * WD) {
    int t = idx >> 8;          // WD == 256
    int c = idx & 255;
    xb[idx] = f2bf(E[(size_t)tok[t] * WD + c]);
  }
}

// ---------------------------------------------------------------------------
// bf16 WMMA GEMM:  C(MxN,f32) = A(MxK,bf16 row-major) @ Bp(packed) [+ bias[N]]
// 128 threads = 4 waves; block tile 64x64; wave owns 16 rows x 64 cols.
// ---------------------------------------------------------------------------
__global__ __launch_bounds__(128) void gemm_bf16_kernel(
    const __bf16* __restrict__ A, const __bf16* __restrict__ Bp,
    const float* __restrict__ bias, float* __restrict__ C,
    int M, int Nn, int K) {
  const int wave = (int)(threadIdx.x >> 5);
  const int row0 = blockIdx.x * 64 + wave * 16;
  const int colb = blockIdx.y * 64;
  const int numNT = Nn >> 4;
  const int ntb = colb >> 4;
  v8f acc[4] = {};
  for (int k0 = 0; k0 < K; k0 += 32) {
    const int kt = k0 >> 5;
    v16bf a = load_a_rm(A, K, row0, k0);
#pragma unroll
    for (int nt = 0; nt < 4; ++nt) {
      v16bf b = load_b_packed(Bp, numNT, kt, ntb + nt);
      acc[nt] = __builtin_amdgcn_wmma_f32_16x16x32_bf16(
          false, a, false, b, (short)0, acc[nt], false, false);
    }
  }
  const int lane = (int)(threadIdx.x & 31);
  const int hi = lane >> 4, nl = lane & 15;
#pragma unroll
  for (int nt = 0; nt < 4; ++nt) {
#pragma unroll
    for (int v = 0; v < 8; ++v) {
      int r = row0 + v + 8 * hi;           // D layout: VGPR v -> row v + 8*hi
      int c = colb + nt * 16 + nl;
      float val = acc[nt][v];
      if (bias) val += bias[c];
      C[(size_t)r * Nn + c] = val;
    }
  }
}

// ---------------------------------------------------------------------------
// Bidirectional LSTM scan. grid=2 (block 0 fwd, block 1 bwd), 1024 threads.
// zx = x@Wx + b precomputed per timestep.  h kept in LDS (f32); bf16 output
// written into concat layout out[t][off + j], off = 0 (fwd) / LD (bwd).
// ---------------------------------------------------------------------------
__global__ __launch_bounds__(1024) void lstm_bidir_kernel(
    const float* __restrict__ zxF, const float* __restrict__ WhF,
    const float* __restrict__ zxB, const float* __restrict__ WhB,
    __bf16* __restrict__ outBf) {
  __shared__ float h_s[LD];
  __shared__ float z_s[GATES];
  const bool bwd = (blockIdx.x == 1);
  const float* zx = bwd ? zxB : zxF;
  const float* Wh = bwd ? WhB : WhF;
  const int off = bwd ? LD : 0;
  const int tid = (int)threadIdx.x;
  float c = 0.0f;
  if (tid < LD) h_s[tid] = 0.0f;
  __syncthreads();
  for (int t = 0; t < N_SEQ; ++t) {
    const int row = bwd ? (N_SEQ - 1 - t) : t;
    float acc = zx[(size_t)row * GATES + tid];
    const float* wcol = Wh + tid;            // column tid, stride GATES
#pragma unroll 8
    for (int k = 0; k < LD; ++k)
      acc = fmaf(h_s[k], wcol[(size_t)k * GATES], acc);
    z_s[tid] = acc;
    __syncthreads();
    if (tid < LD) {
      // Keras gate order [i, f, g, o]
      float ig = sigm(z_s[tid]);
      float fg = sigm(z_s[LD + tid]);
      float gg = fast_tanh(z_s[2 * LD + tid]);
      float og = sigm(z_s[3 * LD + tid]);
      c = fg * c + ig * gg;
      float h = og * fast_tanh(c);
      h_s[tid] = h;
      outBf[(size_t)row * FEAT + off + tid] = f2bf(h);
    }
    __syncthreads();
  }
}

// ---------------------------------------------------------------------------
// Arc scorer: 16x16 pair tile per block, 256 threads.  hidBias is already
// folded into Hh (GEMM bias).  Mh tile + outW staged in LDS (stride 516
// floats: conflict-free AND 16B-aligned rows for float4).
// ---------------------------------------------------------------------------
#define MPAD 516
__global__ __launch_bounds__(256) void arc_kernel(
    const float* __restrict__ Hh, const float* __restrict__ Mh,
    const float* __restrict__ outW, const float* __restrict__ outB,
    float* __restrict__ arc) {
  __shared__ float Mh_s[16 * MPAD];
  __shared__ float w_s[HID];
  const int i0 = blockIdx.x * 16, j0 = blockIdx.y * 16;
  const int tid = (int)threadIdx.x;
  for (int idx = tid; idx < 16 * (HID / 4); idx += 256) {
    int r = idx >> 7, c4 = idx & 127;       // HID/4 == 128
    *reinterpret_cast<float4*>(&Mh_s[r * MPAD + c4 * 4]) =
        *reinterpret_cast<const float4*>(&Mh[(size_t)(j0 + r) * HID + c4 * 4]);
  }
  if (tid < HID / 2) {
    *reinterpret_cast<float2*>(&w_s[tid * 2]) =
        *reinterpret_cast<const float2*>(&outW[tid * 2]);
  }
  __syncthreads();
  const int ii = tid >> 4, jj = tid & 15;
  const float* hrow = Hh + (size_t)(i0 + ii) * HID;   // includes hidBias
  const float* mrow = &Mh_s[jj * MPAD];
  float s = 0.0f;
  for (int h = 0; h < HID; h += 4) {
    float4 hv = *reinterpret_cast<const float4*>(hrow + h);
    float4 mv = *reinterpret_cast<const float4*>(mrow + h);
    float4 wv = *reinterpret_cast<const float4*>(&w_s[h]);
    s = fmaf(fast_tanh(hv.x + mv.x), wv.x, s);
    s = fmaf(fast_tanh(hv.y + mv.y), wv.y, s);
    s = fmaf(fast_tanh(hv.z + mv.z), wv.z, s);
    s = fmaf(fast_tanh(hv.w + mv.w), wv.w, s);
  }
  arc[(size_t)(i0 + ii) * N_SEQ + (j0 + jj)] = s + outB[0];
}

// ---------------------------------------------------------------------------
// Relation scorer: 16x16 pair tile -> M=256 pair rows, N=112 (100 real),
// K=512.  rhidBias already folded into Rh (GEMM bias).  Per 32-wide K chunk:
// 256 threads compute tanh activations into aligned bf16 LDS (row stride 40
// -> conflict-free ds_load_b128), then 8 waves each run 2 M-tiles x 7
// N-tiles of v_wmma_f32_16x16x32_bf16 with packed routW fragments.
// ---------------------------------------------------------------------------
#define ACT_LD 40
__global__ __launch_bounds__(256) void rel_kernel(
    const float* __restrict__ Rh, const float* __restrict__ Rm,
    const __bf16* __restrict__ Wp,   // packed (HID x NRELP)
    const float* __restrict__ routB, float* __restrict__ rel) {
  __shared__ float Rm_s[16 * MPAD];
  __shared__ __attribute__((aligned(16))) __bf16 act_s[256 * ACT_LD];
  const int i0 = blockIdx.x * 16, j0 = blockIdx.y * 16;
  const int tid = (int)threadIdx.x;
  const int wave = tid >> 5;
  for (int idx = tid; idx < 16 * (HID / 4); idx += 256) {
    int r = idx >> 7, c4 = idx & 127;
    *reinterpret_cast<float4*>(&Rm_s[r * MPAD + c4 * 4]) =
        *reinterpret_cast<const float4*>(&Rm[(size_t)(j0 + r) * HID + c4 * 4]);
  }
  __syncthreads();
  const int ii = tid >> 4, jj = tid & 15;
  const float* hrow = Rh + (size_t)(i0 + ii) * HID;   // includes rhidBias
  const float* mrow = &Rm_s[jj * MPAD];
  v8f acc[2][7] = {};
  for (int kc = 0; kc < HID; kc += 32) {
#pragma unroll
    for (int hh = 0; hh < 32; hh += 4) {
      int h = kc + hh;
      float4 hv = *reinterpret_cast<const float4*>(hrow + h);
      float4 mv = *reinterpret_cast<const float4*>(mrow + h);
      v4bf o;
      o[0] = f2bf(fast_tanh(hv.x + mv.x));
      o[1] = f2bf(fast_tanh(hv.y + mv.y));
      o[2] = f2bf(fast_tanh(hv.z + mv.z));
      o[3] = f2bf(fast_tanh(hv.w + mv.w));
      *reinterpret_cast<v4bf*>(&act_s[tid * ACT_LD + hh]) = o;
    }
    __syncthreads();
    const int kt = kc >> 5;
#pragma unroll
    for (int mt = 0; mt < 2; ++mt) {
      v16bf a = load_a_rm(act_s, ACT_LD, (wave * 2 + mt) * 16, 0);
#pragma unroll
      for (int nt = 0; nt < 7; ++nt) {
        v16bf b = load_b_packed(Wp, NRELP >> 4, kt, nt);
        acc[mt][nt] = __builtin_amdgcn_wmma_f32_16x16x32_bf16(
            false, a, false, b, (short)0, acc[mt][nt], false, false);
      }
    }
    __syncthreads();
  }
  const int lane = tid & 31, hi = lane >> 4, nl = lane & 15;
#pragma unroll
  for (int mt = 0; mt < 2; ++mt) {
#pragma unroll
    for (int nt = 0; nt < 7; ++nt) {
#pragma unroll
      for (int v = 0; v < 8; ++v) {
        int m = (wave * 2 + mt) * 16 + v + 8 * hi;   // pair row
        int n = nt * 16 + nl;
        if (n < NREL) {
          int pi = m >> 4, pj = m & 15;
          rel[((size_t)(i0 + pi) * N_SEQ + (j0 + pj)) * NREL + n] =
              acc[mt][nt][v] + routB[n];
        }
      }
    }
  }
}

// ---------------------------------------------------------------------------
extern "C" void kernel_launch(void* const* d_in, const int* in_sizes, int n_in,
                              void* d_out, int out_size, void* d_ws, size_t ws_size,
                              hipStream_t stream) {
  (void)in_sizes; (void)n_in; (void)out_size; (void)ws_size;

  const int*   tokens   = (const int*)  d_in[0];
  const float* E        = (const float*)d_in[1];
  const float* Wx_f1    = (const float*)d_in[2];
  const float* Wh_f1    = (const float*)d_in[3];
  const float* b_f1     = (const float*)d_in[4];
  const float* Wx_b1    = (const float*)d_in[5];
  const float* Wh_b1    = (const float*)d_in[6];
  const float* b_b1     = (const float*)d_in[7];
  const float* Wx_f2    = (const float*)d_in[8];
  const float* Wh_f2    = (const float*)d_in[9];
  const float* b_f2     = (const float*)d_in[10];
  const float* Wx_b2    = (const float*)d_in[11];
  const float* Wh_b2    = (const float*)d_in[12];
  const float* b_b2     = (const float*)d_in[13];
  const float* hidFOH   = (const float*)d_in[14];
  const float* hidFOM   = (const float*)d_in[15];
  const float* hidBias  = (const float*)d_in[16];
  const float* outW     = (const float*)d_in[17];
  const float* outB     = (const float*)d_in[18];
  const float* rhidFOH  = (const float*)d_in[19];
  const float* rhidFOM  = (const float*)d_in[20];
  const float* rhidBias = (const float*)d_in[21];
  const float* routW    = (const float*)d_in[22];
  const float* routB    = (const float*)d_in[23];

  float* arc = (float*)d_out;
  float* rel = arc + (size_t)N_SEQ * N_SEQ;

  // ---- carve workspace (~19 MB) ----
  char* base = (char*)d_ws;
  size_t off = 0;
  auto alloc = [&](size_t bytes) -> void* {
    off = (off + 255) & ~(size_t)255;
    void* p = base + off;
    off += bytes;
    return p;
  };
  __bf16* xb     = (__bf16*)alloc((size_t)N_SEQ * WD * 2);
  __bf16* WxF1p  = (__bf16*)alloc((size_t)WD * GATES * 2);
  __bf16* WxB1p  = (__bf16*)alloc((size_t)WD * GATES * 2);
  __bf16* WxF2p  = (__bf16*)alloc((size_t)FEAT * GATES * 2);
  __bf16* WxB2p  = (__bf16*)alloc((size_t)FEAT * GATES * 2);
  __bf16* FOHp   = (__bf16*)alloc((size_t)FEAT * HID * 2);
  __bf16* FOMp   = (__bf16*)alloc((size_t)FEAT * HID * 2);
  __bf16* RFOHp  = (__bf16*)alloc((size_t)FEAT * HID * 2);
  __bf16* RFOMp  = (__bf16*)alloc((size_t)FEAT * HID * 2);
  __bf16* routWp = (__bf16*)alloc((size_t)HID * NRELP * 2);
  float*  zxF1   = (float*) alloc((size_t)N_SEQ * GATES * 4);
  float*  zxB1   = (float*) alloc((size_t)N_SEQ * GATES * 4);
  float*  zxF2   = (float*) alloc((size_t)N_SEQ * GATES * 4);
  float*  zxB2   = (float*) alloc((size_t)N_SEQ * GATES * 4);
  __bf16* h1b    = (__bf16*)alloc((size_t)N_SEQ * FEAT * 2);
  __bf16* h2b    = (__bf16*)alloc((size_t)N_SEQ * FEAT * 2);
  float*  Hh     = (float*) alloc((size_t)N_SEQ * HID * 4);
  float*  Mh     = (float*) alloc((size_t)N_SEQ * HID * 4);
  float*  Rh     = (float*) alloc((size_t)N_SEQ * HID * 4);
  float*  Rm     = (float*) alloc((size_t)N_SEQ * HID * 4);

  const int T = 256;
  auto pack = [&](const float* s, __bf16* d, int K, int Npad, int Nsrc) {
    int n = K * Npad;
    pack_b_kernel<<<(n + T - 1) / T, T, 0, stream>>>(s, d, K, Npad, Nsrc);
  };
  pack(Wx_f1, WxF1p, WD, GATES, GATES);
  pack(Wx_b1, WxB1p, WD, GATES, GATES);
  pack(Wx_f2, WxF2p, FEAT, GATES, GATES);
  pack(Wx_b2, WxB2p, FEAT, GATES, GATES);
  pack(hidFOH, FOHp, FEAT, HID, HID);
  pack(hidFOM, FOMp, FEAT, HID, HID);
  pack(rhidFOH, RFOHp, FEAT, HID, HID);
  pack(rhidFOM, RFOMp, FEAT, HID, HID);
  pack(routW, routWp, HID, NRELP, NREL);

  embed_kernel<<<(N_SEQ * WD + T - 1) / T, T, 0, stream>>>(tokens, E, xb);

  // layer-1 input projections: (512x256)@(256x1024)+b
  gemm_bf16_kernel<<<dim3(N_SEQ / 64, GATES / 64), 128, 0, stream>>>(
      xb, WxF1p, b_f1, zxF1, N_SEQ, GATES, WD);
  gemm_bf16_kernel<<<dim3(N_SEQ / 64, GATES / 64), 128, 0, stream>>>(
      xb, WxB1p, b_b1, zxB1, N_SEQ, GATES, WD);

  lstm_bidir_kernel<<<2, 1024, 0, stream>>>(zxF1, Wh_f1, zxB1, Wh_b1, h1b);

  // layer-2 input projections: (512x512)@(512x1024)+b
  gemm_bf16_kernel<<<dim3(N_SEQ / 64, GATES / 64), 128, 0, stream>>>(
      h1b, WxF2p, b_f2, zxF2, N_SEQ, GATES, FEAT);
  gemm_bf16_kernel<<<dim3(N_SEQ / 64, GATES / 64), 128, 0, stream>>>(
      h1b, WxB2p, b_b2, zxB2, N_SEQ, GATES, FEAT);

  lstm_bidir_kernel<<<2, 1024, 0, stream>>>(zxF2, Wh_f2, zxB2, Wh_b2, h2b);

  // scorer projections: (512x512)@(512x512); fold the pre-tanh biases into
  // the "head" side so the scorer kernels only stream two operands.
  gemm_bf16_kernel<<<dim3(N_SEQ / 64, HID / 64), 128, 0, stream>>>(
      h2b, FOHp, hidBias, Hh, N_SEQ, HID, FEAT);
  gemm_bf16_kernel<<<dim3(N_SEQ / 64, HID / 64), 128, 0, stream>>>(
      h2b, FOMp, nullptr, Mh, N_SEQ, HID, FEAT);
  gemm_bf16_kernel<<<dim3(N_SEQ / 64, HID / 64), 128, 0, stream>>>(
      h2b, RFOHp, rhidBias, Rh, N_SEQ, HID, FEAT);
  gemm_bf16_kernel<<<dim3(N_SEQ / 64, HID / 64), 128, 0, stream>>>(
      h2b, RFOMp, nullptr, Rm, N_SEQ, HID, FEAT);

  // fused pairwise scorers
  arc_kernel<<<dim3(N_SEQ / 16, N_SEQ / 16), 256, 0, stream>>>(
      Hh, Mh, outW, outB, arc);
  rel_kernel<<<dim3(N_SEQ / 16, N_SEQ / 16), 256, 0, stream>>>(
      Rh, Rm, routWp, routB, rel);
}
